// BertBackbone_11012296147148
// MI455X (gfx1250) — matrix operand order repaired and checked by reading, not compiled, not run on hardware
//
#include <hip/hip_runtime.h>

// ---------------------------------------------------------------------------
// Model constants (fixed by the reference)
// ---------------------------------------------------------------------------
#define BB   32
#define SS   512
#define HH   768
#define NHH  12
#define DHH  64
#define II   3072
#define LL   6
#define MM   (BB * SS)          // 16384 rows

typedef __attribute__((ext_vector_type(16))) __bf16 v16bf;
typedef __attribute__((ext_vector_type(8)))  float  v8f;

union Frag16 { v16bf v; unsigned int u[8]; };

__device__ __forceinline__ v8f wmma_bf16(v16bf a, v16bf b, v8f c) {
  return __builtin_amdgcn_wmma_f32_16x16x32_bf16(false, a, false, b,
                                                 (short)0, c, false, false);
}

// fp32 -> bf16 bits, round-to-nearest-even
__device__ __forceinline__ unsigned short f2bf(float f) {
  unsigned int u = __float_as_uint(f);
  unsigned int r = (u + 0x7FFFu + ((u >> 16) & 1u)) >> 16;
  return (unsigned short)r;
}

// CDNA5 async global->LDS copy (ASYNCcnt-tracked, no VGPR round trip).
template <int OFF>
__device__ __forceinline__ void async_b128(unsigned lds_addr, const void* gaddr) {
  asm volatile("global_load_async_to_lds_b128 %0, %1, off offset:%c2"
               :: "v"(lds_addr), "v"(gaddr), "i"(OFF) : "memory");
}
template <int N>
__device__ __forceinline__ void wait_async_le() {
  asm volatile("s_wait_asynccnt %c0" :: "i"(N) : "memory");
}
__device__ __forceinline__ unsigned lds_off(const void* p) {
  return (unsigned)(uintptr_t)p;
}

__device__ __forceinline__ bool mask_ok(int q, int k, int n, int us) {
  const bool xq = q >= n, xk = k >= n;
  const int uq = xq ? (q - n) / us : q / us;
  const int uk = xk ? (k - n) / us : k / us;
  return ((uq == uk) && (xq == xk)) ||
         ((uq >  uk) && xk && !xq)  ||
         ((uq >= uk) && xk &&  xq);
}

// ---------------------------------------------------------------------------
// fp32 -> bf16 convert + transpose (one-time): dst[l][n][k] = bf16(src[l][k][n])
// ---------------------------------------------------------------------------
__global__ __launch_bounds__(256) void f2bfT_kernel(const float* __restrict__ src,
                                                    unsigned short* __restrict__ dst,
                                                    int K, int N) {
  const float* s = src + (size_t)blockIdx.y * K * N;
  unsigned short* d = dst + (size_t)blockIdx.y * K * N;
  const int idx = blockIdx.x * 256 + threadIdx.x;   // over K*N, coalesced read
  if (idx < K * N) {
    const int k = idx / N, n = idx - k * N;
    d[(size_t)n * K + k] = f2bf(s[idx]);
  }
}

// ---------------------------------------------------------------------------
// Embedding gather: out[row] = word_emb[id] + pos_emb[s] + type_emb[0]
// ---------------------------------------------------------------------------
__global__ __launch_bounds__(256) void embed_kernel(const int* __restrict__ ids,
                                                    const float* __restrict__ we,
                                                    const float* __restrict__ pe,
                                                    const float* __restrict__ te,
                                                    float* __restrict__ out) {
  const int row = blockIdx.x;
  const int id  = ids[row];
  const int s   = row & (SS - 1);
  #pragma unroll
  for (int i = 0; i < 3; i++) {
    const int c = threadIdx.x + i * 256;
    out[(size_t)row * HH + c] =
        we[(size_t)id * HH + c] + pe[(size_t)s * HH + c] + te[c];
  }
}

// ---------------------------------------------------------------------------
// LayerNorm: outf/outb = LN(X + R) * g + b   (R / outb may be null).
// ---------------------------------------------------------------------------
__global__ __launch_bounds__(256) void ln_kernel(const float* __restrict__ X,
                                                 const float* __restrict__ R,
                                                 const float* __restrict__ g,
                                                 const float* __restrict__ bt,
                                                 float* __restrict__ outf,
                                                 unsigned short* __restrict__ outb) {
  const int wave = threadIdx.x >> 5;
  const int lane = threadIdx.x & 31;
  const int row  = blockIdx.x * 8 + wave;
  const float* x = X + (size_t)row * HH;
  const float* r = R ? R + (size_t)row * HH : nullptr;

  float v[24];
  float sum = 0.f;
  #pragma unroll
  for (int i = 0; i < 24; i++) {
    const int c = lane + i * 32;
    float t = x[c] + (r ? r[c] : 0.f);
    v[i] = t;
    sum += t;
  }
  #pragma unroll
  for (int off = 1; off < 32; off <<= 1) sum += __shfl_xor(sum, off, 32);
  const float mu = sum * (1.0f / HH);
  float vs = 0.f;
  #pragma unroll
  for (int i = 0; i < 24; i++) { float d = v[i] - mu; vs += d * d; }
  #pragma unroll
  for (int off = 1; off < 32; off <<= 1) vs += __shfl_xor(vs, off, 32);
  const float rstd = rsqrtf(vs * (1.0f / HH) + 1e-12f);
  #pragma unroll
  for (int i = 0; i < 24; i++) {
    const int c = lane + i * 32;
    const float o = (v[i] - mu) * rstd * g[c] + bt[c];
    outf[(size_t)row * HH + c] = o;
    if (outb) outb[(size_t)row * HH + c] = f2bf(o);
  }
}

// ---------------------------------------------------------------------------
// WMMA GEMM:  C[M,N] = act(A_bf16[M,K] * WT_bf16[N,K]^T + bias)
//   mode 0: store fp32 row-major
//   mode 1: store bf16 scattered to [B, NH, S, DH]   (Q/K projections)
//   mode 2: exact-erf GELU, store bf16 row-major
//   mode 3: store bf16 scattered to [B, NH, DH, S]   (V projection, transposed)
// Block: 128 threads (4 waves), 64x128 tile; each wave 32x64 (2x4 WMMA tiles).
// Both tiles staged via async global->LDS, double-buffered on ASYNCcnt,
// last iteration peeled for a branch-free steady state.
// ---------------------------------------------------------------------------
__global__ __launch_bounds__(128) void gemm_bf16_kernel(
    const unsigned short* __restrict__ A, const unsigned short* __restrict__ WT,
    const float* __restrict__ bias, float* __restrict__ Cf,
    unsigned short* __restrict__ Cb, int M, int N, int K, int mode) {
  __shared__ unsigned short As[2][64][32];    // [buf][m][k], k pairs contiguous
  __shared__ unsigned short Bs[2][128][32];   // [buf][n][k], k pairs contiguous

  const int tid  = threadIdx.x;
  const int lane = tid & 31;
  const int wave = tid >> 5;
  const int l15  = lane & 15;
  const int hl   = lane >> 4;
  const int wm   = (wave >> 1) * 32;   // 0 / 32
  const int wn   = (wave & 1) * 64;    // 0 / 64
  const int mb   = blockIdx.y * 64;
  const int nb   = blockIdx.x * 128;

  v8f cacc[2][4] = {};

  const int ar = tid >> 1;          // A row 0..63
  const int ac = (tid & 1) * 16;    // A halves 0/16 (32B per thread)
  const unsigned short* aptr = A  + (size_t)(mb + ar) * K + ac;
  const unsigned short* bptr = WT + (size_t)(nb + tid) * K;   // B row tid (64B)
  const unsigned ldsA[2] = { lds_off(&As[0][ar][ac]), lds_off(&As[1][ar][ac]) };
  const unsigned ldsB[2] = { lds_off(&Bs[0][tid][0]), lds_off(&Bs[1][tid][0]) };

  auto issue = [&](int buf, int kc) {
    const unsigned short* sa = aptr + kc;
    async_b128<0>(ldsA[buf], sa);
    async_b128<16>(ldsA[buf], sa);
    const unsigned short* sb = bptr + kc;
    async_b128<0>(ldsB[buf], sb);
    async_b128<16>(ldsB[buf], sb);
    async_b128<32>(ldsB[buf], sb);
    async_b128<48>(ldsB[buf], sb);
  };
  auto compute = [&](int cur) {
    Frag16 af[2], bfr[4];
    #pragma unroll
    for (int i = 0; i < 2; i++) {
      const unsigned int* ap = (const unsigned int*)&As[cur][wm + i * 16 + l15][0];
      #pragma unroll
      for (int v = 0; v < 4; v++) {
        af[i].u[v]     = ap[hl * 4 + v];       // k = hl*8 + 2v
        af[i].u[4 + v] = ap[8 + hl * 4 + v];   // k = 16 + hl*8 + 2v
      }
    }
    #pragma unroll
    for (int j = 0; j < 4; j++) {
      const unsigned int* bp = (const unsigned int*)&Bs[cur][wn + j * 16 + l15][0];
      #pragma unroll
      for (int v = 0; v < 8; v++) bfr[j].u[v] = bp[(v >> 2) * 8 + hl * 4 + (v & 3)];  // k = (v>>2)*16 + hl*8 + 2(v&3)
    }
    #pragma unroll
    for (int i = 0; i < 2; i++)
      #pragma unroll
      for (int j = 0; j < 4; j++)
        cacc[i][j] = wmma_bf16(af[i].v, bfr[j].v, cacc[i][j]);
  };

  // prologue: stage tile 0
  issue(0, 0);
  const int T = K >> 5;
  for (int t = 0; t < T - 1; t++) {
    issue((t + 1) & 1, (t + 1) * 32);
    wait_async_le<6>();   // tile t resident (in-order retirement)
    __syncthreads();
    compute(t & 1);
    __syncthreads();      // all waves done reading buf before refill
  }
  wait_async_le<0>();
  __syncthreads();
  compute((T - 1) & 1);

  // ---- epilogue
  #pragma unroll
  for (int i = 0; i < 2; i++) {
    #pragma unroll
    for (int j = 0; j < 4; j++) {
      const int col = nb + wn + j * 16 + l15;
      const float bv = bias[col];
      #pragma unroll
      for (int r = 0; r < 8; r++) {
        const int row = mb + wm + i * 16 + hl * 8 + r;
        float val = cacc[i][j][r] + bv;
        if (mode == 0) {
          Cf[(size_t)row * N + col] = val;
        } else if (mode == 1) {
          const int bidx = row >> 9, s = row & (SS - 1);
          const int hh = col >> 6, d = col & (DHH - 1);
          Cb[(((size_t)bidx * NHH + hh) * SS + s) * DHH + d] = f2bf(val);
        } else if (mode == 3) {
          const int bidx = row >> 9, s = row & (SS - 1);
          const int hh = col >> 6, d = col & (DHH - 1);
          Cb[(((size_t)bidx * NHH + hh) * DHH + d) * SS + s] = f2bf(val);
        } else {
          val = 0.5f * val * (1.0f + erff(val * 0.70710678118654752f));
          Cb[(size_t)row * N + col] = f2bf(val);
        }
      }
    }
  }
}

// ---------------------------------------------------------------------------
// Flash attention: block = 128 threads (4 waves), 64 q rows per block.
// grid = (S/64, NH, B). Q/K bf16 [B,NH,S,DH]; V bf16 [B,NH,DH,S] (transposed).
// Out: ctx bf16 [B,S,H]. All three stagings are direct async copies.
// ---------------------------------------------------------------------------
__global__ __launch_bounds__(128) void attn_kernel(
    const unsigned short* __restrict__ Q, const unsigned short* __restrict__ Kk,
    const unsigned short* __restrict__ V, unsigned short* __restrict__ ctx,
    const int* __restrict__ np, const int* __restrict__ usp) {
  __shared__ unsigned short Qs[64][64];     // [qrow][d]
  __shared__ unsigned short Kc[64][64];     // [kv][d]  (chunk)
  __shared__ unsigned short Vt[64][64];     // [d][kv]  (chunk)
  __shared__ unsigned short Ps[4][16][32];  // per-wave P tile (k=32)

  const int tid  = threadIdx.x;
  const int lane = tid & 31;
  const int wave = tid >> 5;
  const int l15  = lane & 15;
  const int hl   = lane >> 4;
  const int b  = blockIdx.z;
  const int h  = blockIdx.y;
  const int qb = blockIdx.x * 64;
  const int nmask = np[0];
  const int us    = usp[0];
  const size_t headoff = ((size_t)b * NHH + h) * SS * DHH;

  const int sr = tid >> 1;           // staging row 0..63
  const int sc = (tid & 1) * 32;     // staging halves 0/32 (64B per thread)

  // ---- stage Q (async: 64 rows x 64 d)
  {
    const unsigned short* src = Q + headoff + (size_t)(qb + sr) * DHH + sc;
    const unsigned ldsQ = lds_off(&Qs[sr][sc]);
    async_b128<0>(ldsQ, src);
    async_b128<16>(ldsQ, src);
    async_b128<32>(ldsQ, src);
    async_b128<48>(ldsQ, src);
  }
  wait_async_le<0>();
  __syncthreads();

  // ---- Q fragments (hoisted; 2 k-chunks of DH=64)
  Frag16 qf[2];
  {
    const unsigned int* qp = (const unsigned int*)&Qs[wave * 16 + l15][0];
    #pragma unroll
    for (int cc = 0; cc < 2; cc++) {
      #pragma unroll
      for (int v = 0; v < 4; v++) {
        qf[cc].u[v]     = qp[cc * 16 + hl * 4 + v];
        qf[cc].u[4 + v] = qp[cc * 16 + 8 + hl * 4 + v];
      }
    }
  }

  float mrow[8], lrow[8];
  #pragma unroll
  for (int r = 0; r < 8; r++) { mrow[r] = -1e30f; lrow[r] = 0.f; }
  v8f acc[4] = {};

  const unsigned ldsK = lds_off(&Kc[sr][sc]);
  const unsigned ldsV = lds_off(&Vt[sr][sc]);

  for (int kv0 = 0; kv0 < SS; kv0 += 64) {
    __syncthreads();
    // ---- stage K chunk (async, [kv][d]) and V chunk (async, [d][kv])
    {
      const unsigned short* sk = Kk + headoff + (size_t)(kv0 + sr) * DHH + sc;
      async_b128<0>(ldsK, sk);
      async_b128<16>(ldsK, sk);
      async_b128<32>(ldsK, sk);
      async_b128<48>(ldsK, sk);
      const unsigned short* sv = V + headoff + (size_t)sr * SS + kv0 + sc;
      async_b128<0>(ldsV, sv);
      async_b128<16>(ldsV, sv);
      async_b128<32>(ldsV, sv);
      async_b128<48>(ldsV, sv);
    }
    wait_async_le<0>();
    __syncthreads();

    #pragma unroll
    for (int sub = 0; sub < 2; sub++) {
      const int kvbase = kv0 + sub * 32;
      float s0[8], s1[8];
      // ---- scores: two 16x16 tiles via QK^T WMMA (k = DH = 64 -> 2 chunks)
      #pragma unroll
      for (int j = 0; j < 2; j++) {
        const unsigned int* kp =
            (const unsigned int*)&Kc[sub * 32 + j * 16 + l15][0];
        Frag16 kb0, kb1;
        #pragma unroll
        for (int v = 0; v < 8; v++) {
          kb0.u[v] = kp[(v >> 2) * 8 + hl * 4 + (v & 3)];
          kb1.u[v] = kp[16 + (v >> 2) * 8 + hl * 4 + (v & 3)];
        }
        v8f scv = {};
        scv = wmma_bf16(qf[0].v, kb0.v, scv);
        scv = wmma_bf16(qf[1].v, kb1.v, scv);
        const int col = kvbase + j * 16 + l15;
        float* so = j ? s1 : s0;
        #pragma unroll
        for (int r = 0; r < 8; r++) {
          const int qrow = qb + wave * 16 + hl * 8 + r;
          float sv = scv[r] * 0.125f;   // 1/sqrt(64)
          so[r] = mask_ok(qrow, col, nmask, us) ? sv : -1e30f;
        }
      }
      // ---- online softmax update (row reductions across the 16-lane group)
      float fr[8], p0[8], p1[8];
      #pragma unroll
      for (int r = 0; r < 8; r++) {
        float mx = fmaxf(s0[r], s1[r]);
        #pragma unroll
        for (int off = 1; off < 16; off <<= 1)
          mx = fmaxf(mx, __shfl_xor(mx, off, 16));
        const float mnew = fmaxf(mrow[r], mx);
        fr[r] = expf(mrow[r] - mnew);
        p0[r] = (s0[r] > -1e29f) ? expf(s0[r] - mnew) : 0.f;
        p1[r] = (s1[r] > -1e29f) ? expf(s1[r] - mnew) : 0.f;
        float ps = p0[r] + p1[r];
        #pragma unroll
        for (int off = 1; off < 16; off <<= 1) ps += __shfl_xor(ps, off, 16);
        lrow[r] = lrow[r] * fr[r] + ps;
        mrow[r] = mnew;
      }
      #pragma unroll
      for (int jn = 0; jn < 4; jn++)
        #pragma unroll
        for (int r = 0; r < 8; r++) acc[jn][r] *= fr[r];
      // ---- write P tile to LDS in C-layout, re-gather as A fragment
      #pragma unroll
      for (int r = 0; r < 8; r++) {
        Ps[wave][hl * 8 + r][l15]      = f2bf(p0[r]);
        Ps[wave][hl * 8 + r][16 + l15] = f2bf(p1[r]);
      }
      __syncthreads();
      Frag16 pf;
      {
        const unsigned int* pp = (const unsigned int*)&Ps[wave][l15][0];
        #pragma unroll
        for (int v = 0; v < 4; v++) {
          pf.u[v]     = pp[hl * 4 + v];
          pf.u[4 + v] = pp[8 + hl * 4 + v];
        }
      }
      // ---- ctx += P * V  (4 n-tiles over DH)
      #pragma unroll
      for (int jn = 0; jn < 4; jn++) {
        const unsigned int* vp = (const unsigned int*)&Vt[jn * 16 + l15][0];
        Frag16 vb;
        #pragma unroll
        for (int v = 0; v < 8; v++) vb.u[v] = vp[sub * 16 + (v >> 2) * 8 + hl * 4 + (v & 3)];
        acc[jn] = wmma_bf16(pf.v, vb.v, acc[jn]);
      }
      __syncthreads();
    }
  }
  // ---- normalize & scatter ctx (bf16) -> [B, S, H]
  #pragma unroll
  for (int jn = 0; jn < 4; jn++) {
    #pragma unroll
    for (int r = 0; r < 8; r++) {
      const int qrow = qb + wave * 16 + hl * 8 + r;
      const float l = lrow[r];
      const float val = (l > 0.f) ? acc[jn][r] / l : 0.f;
      ctx[((size_t)b * SS + qrow) * HH + h * DHH + jn * 16 + l15] = f2bf(val);
    }
  }
}

// ---------------------------------------------------------------------------
// Host launcher
// ---------------------------------------------------------------------------
extern "C" void kernel_launch(void* const* d_in, const int* in_sizes, int n_in,
                              void* d_out, int out_size, void* d_ws,
                              size_t ws_size, hipStream_t stream) {
  if (n_in < 24) return;
  // Detect input flattening order: jax pytree flatten sorts dict keys,
  // which puts scalar 'n' at index 1. Otherwise assume insertion order.
  int i_ids, i_we, i_pe, i_te, i_eg, i_eb;
  int i_Wq, i_bq, i_Wk, i_bk, i_Wv, i_bv, i_Wo, i_bo;
  int i_l1g, i_l1b, i_Wi, i_bi, i_Wf, i_bf, i_l2g, i_l2b, i_us, i_n;
  if (in_sizes[1] == 1) {   // sorted-key pytree order
    i_ids = 0;  i_n = 1;   i_eb = 2;  i_eg = 3;
    i_Wf = 4;   i_Wi = 5;  i_Wk = 6;  i_Wo = 7;  i_Wq = 8;  i_Wv = 9;
    i_bf = 10;  i_bi = 11; i_bk = 12; i_bo = 13; i_bq = 14; i_bv = 15;
    i_l1b = 16; i_l1g = 17; i_l2b = 18; i_l2g = 19;
    i_pe = 20;  i_te = 21; i_we = 22; i_us = 23;
  } else {                  // dict insertion order
    i_ids = 0;  i_we = 1;  i_pe = 2;  i_te = 3;  i_eg = 4;  i_eb = 5;
    i_Wq = 6;   i_bq = 7;  i_Wk = 8;  i_bk = 9;  i_Wv = 10; i_bv = 11;
    i_Wo = 12;  i_bo = 13; i_l1g = 14; i_l1b = 15;
    i_Wi = 16;  i_bi = 17; i_Wf = 18; i_bf = 19; i_l2g = 20; i_l2b = 21;
    i_us = 22;  i_n = 23;
  }

  const int*   ids = (const int*)d_in[i_ids];
  const float* we  = (const float*)d_in[i_we];
  const float* pe  = (const float*)d_in[i_pe];
  const float* te  = (const float*)d_in[i_te];
  const float* eg  = (const float*)d_in[i_eg];
  const float* eb  = (const float*)d_in[i_eb];
  const float* Wq = (const float*)d_in[i_Wq], *Bq = (const float*)d_in[i_bq];
  const float* Wk = (const float*)d_in[i_Wk], *Bk = (const float*)d_in[i_bk];
  const float* Wv = (const float*)d_in[i_Wv], *Bv = (const float*)d_in[i_bv];
  const float* Wo = (const float*)d_in[i_Wo], *Bo = (const float*)d_in[i_bo];
  const float* Wi = (const float*)d_in[i_Wi], *Bi = (const float*)d_in[i_bi];
  const float* Wf = (const float*)d_in[i_Wf], *Bf = (const float*)d_in[i_bf];
  const float* L1g = (const float*)d_in[i_l1g], *L1b = (const float*)d_in[i_l1b];
  const float* L2g = (const float*)d_in[i_l2g], *L2b = (const float*)d_in[i_l2b];
  const int* np  = (const int*)d_in[i_n];
  const int* usp = (const int*)d_in[i_us];

  // ---- workspace layout
  char* ws = (char*)d_ws;
  size_t off = 0;
  auto take = [&](size_t bytes) -> char* {
    char* p = ws + off;
    off += (bytes + 255) & ~(size_t)255;
    return p;
  };
  const size_t headElems = (size_t)BB * NHH * SS * DHH;   // 12.58M
  unsigned short* qbuf = (unsigned short*)take(headElems * 2);
  unsigned short* kbuf = (unsigned short*)take(headElems * 2);
  unsigned short* vbuf = (unsigned short*)take(headElems * 2);
  const size_t whh = (size_t)LL * HH * HH;
  const size_t whi = (size_t)LL * HH * II;
  unsigned short* wq_bf = (unsigned short*)take(whh * 2);   // transposed [N,K]
  unsigned short* wk_bf = (unsigned short*)take(whh * 2);
  unsigned short* wv_bf = (unsigned short*)take(whh * 2);
  unsigned short* wo_bf = (unsigned short*)take(whh * 2);
  unsigned short* wi_bf = (unsigned short*)take(whi * 2);
  unsigned short* wf_bf = (unsigned short*)take(whi * 2);
  float*          xbuf = (float*)take((size_t)MM * HH * 4);
  unsigned short* xbf  = (unsigned short*)take((size_t)MM * HH * 2);
  float*          ybuf = (float*)take((size_t)MM * HH * 4);
  float*          cbuf = (float*)take((size_t)MM * HH * 4);
  unsigned short* cbf  = (unsigned short*)take((size_t)MM * HH * 2);  // ctx/x1 bf16
  unsigned short* hbf  = (unsigned short*)take((size_t)MM * II * 2);
  (void)ws_size;

  // ---- one-shot weight convert + transpose (idempotent, deterministic)
  auto convT = [&](const float* s, unsigned short* d, int K, int N) {
    dim3 grid((K * N + 255) / 256, LL);
    f2bfT_kernel<<<grid, 256, 0, stream>>>(s, d, K, N);
  };
  convT(Wq, wq_bf, HH, HH); convT(Wk, wk_bf, HH, HH); convT(Wv, wv_bf, HH, HH);
  convT(Wo, wo_bf, HH, HH); convT(Wi, wi_bf, HH, II); convT(Wf, wf_bf, II, HH);

  // ---- embeddings + LN
  embed_kernel<<<MM, 256, 0, stream>>>(ids, we, pe, te, ybuf);
  ln_kernel<<<MM / 8, 256, 0, stream>>>(ybuf, nullptr, eg, eb, xbuf, xbf);

  const dim3 gH(HH / 128, MM / 64);   // (6, 256)
  const dim3 gI(II / 128, MM / 64);   // (24, 256)
  const dim3 gA(SS / 64, NHH, BB);    // (8, 12, 32)

  for (int l = 0; l < LL; l++) {
    const unsigned short* wql = wq_bf + (size_t)l * HH * HH;
    const unsigned short* wkl = wk_bf + (size_t)l * HH * HH;
    const unsigned short* wvl = wv_bf + (size_t)l * HH * HH;
    const unsigned short* wol = wo_bf + (size_t)l * HH * HH;
    const unsigned short* wil = wi_bf + (size_t)l * HH * II;
    const unsigned short* wfl = wf_bf + (size_t)l * II * HH;
    const float* bq = Bq + (size_t)l * HH;
    const float* bk = Bk + (size_t)l * HH;
    const float* bv = Bv + (size_t)l * HH;
    const float* bo = Bo + (size_t)l * HH;
    const float* bi = Bi + (size_t)l * II;
    const float* bf = Bf + (size_t)l * HH;

    // QKV projections: Q/K -> [B,NH,S,DH], V -> [B,NH,DH,S]
    gemm_bf16_kernel<<<gH, 128, 0, stream>>>(xbf, wql, bq, nullptr, qbuf,
                                             MM, HH, HH, 1);
    gemm_bf16_kernel<<<gH, 128, 0, stream>>>(xbf, wkl, bk, nullptr, kbuf,
                                             MM, HH, HH, 1);
    gemm_bf16_kernel<<<gH, 128, 0, stream>>>(xbf, wvl, bv, nullptr, vbuf,
                                             MM, HH, HH, 3);
    // flash attention -> ctx bf16 [B,S,H]
    attn_kernel<<<gA, 128, 0, stream>>>(qbuf, kbuf, vbuf, cbf, np, usp);
    // output projection + bias (fp32 out for the residual+LN)
    gemm_bf16_kernel<<<gH, 128, 0, stream>>>(cbf, wol, bo, ybuf, nullptr,
                                             MM, HH, HH, 0);
    // x1 = LN(proj + x)  -> fp32 (residual) + bf16 (GEMM A)
    ln_kernel<<<MM / 8, 256, 0, stream>>>(ybuf, xbuf, L1g + (size_t)l * HH,
                                          L1b + (size_t)l * HH, cbuf, cbf);
    // FFN up + GELU -> bf16
    gemm_bf16_kernel<<<gI, 128, 0, stream>>>(cbf, wil, bi, nullptr, hbf,
                                             MM, II, HH, 2);
    // FFN down + bias -> fp32
    gemm_bf16_kernel<<<gH, 128, 0, stream>>>(hbf, wfl, bf, ybuf, nullptr,
                                             MM, HH, II, 0);
    // x = LN(ffn + x1) -> fp32 + bf16
    ln_kernel<<<MM / 8, 256, 0, stream>>>(ybuf, cbuf, L2g + (size_t)l * HH,
                                          L2b + (size_t)l * HH, xbuf, xbf);
  }

  hipMemcpyAsync(d_out, xbuf, (size_t)MM * HH * sizeof(float),
                 hipMemcpyDeviceToDevice, stream);
}
